// _LlamaDecoderLayer_26199300505690
// MI455X (gfx1250) — matrix operand order repaired and checked by reading, not compiled, not run on hardware
//
#include <hip/hip_runtime.h>
#include <hip/hip_bf16.h>
#include <math.h>

static constexpr int kT      = 2048;
static constexpr int kHidden = 4096;
static constexpr int kNH     = 32;
static constexpr int kNKV    = 8;
static constexpr int kHD     = 128;
static constexpr int kInter  = 14336;

typedef __attribute__((ext_vector_type(16))) __bf16        v16bf;
typedef __attribute__((ext_vector_type(8)))  float         v8f;
typedef __attribute__((ext_vector_type(4)))  unsigned int  u32x4;
typedef __attribute__((ext_vector_type(4)))  float         f32x4;

union FragBF { u32x4 q[2]; v16bf v; };

__device__ __forceinline__ unsigned short f2bf(float f) {
  union { float f; unsigned u; } x; x.f = f;
  unsigned r = x.u + 0x7FFFu + ((x.u >> 16) & 1u);   // round-to-nearest-even
  return (unsigned short)(r >> 16);
}

__device__ __forceinline__ unsigned pack_bf2(float lo, float hi) {
  return (unsigned)f2bf(lo) | ((unsigned)f2bf(hi) << 16);
}

__device__ __forceinline__ void wait_lds() {
#if __has_builtin(__builtin_amdgcn_s_wait_dscnt)
  __builtin_amdgcn_s_wait_dscnt(0);
#else
  __asm__ volatile("s_wait_dscnt 0" ::: "memory");
#endif
}

// ---------------------------------------------------------------- rmsnorm -> bf16
__global__ __launch_bounds__(256)
void rmsnorm_kernel(const float* __restrict__ x, const float* __restrict__ w,
                    unsigned short* __restrict__ out) {
  const int row = blockIdx.x;
  const int tid = threadIdx.x;
  const float* xr = x + (size_t)row * kHidden;
  float ss = 0.f;
  for (int i = tid; i < kHidden; i += 256) { float v = xr[i]; ss += v * v; }
#pragma unroll
  for (int d = 1; d < 32; d <<= 1) ss += __shfl_xor(ss, d, 32);
  __shared__ float red[8];
  if ((tid & 31) == 0) red[tid >> 5] = ss;
  __syncthreads();
  float tot = 0.f;
#pragma unroll
  for (int i = 0; i < 8; ++i) tot += red[i];
  const float inv = rsqrtf(tot / (float)kHidden + 1e-5f);
  for (int i = tid; i < kHidden; i += 256)
    out[(size_t)row * kHidden + i] = f2bf(xr[i] * inv * w[i]);
}

// ---------------------------------------------------- rope(q,k) + cvt(v) -> bf16
__global__ __launch_bounds__(256)
void rope_cvt_kernel(const float* __restrict__ qf, const float* __restrict__ kf,
                     const float* __restrict__ vf, const int* __restrict__ pos,
                     unsigned short* __restrict__ qb, unsigned short* __restrict__ kb,
                     unsigned short* __restrict__ vb) {
  const int t = blockIdx.x, tid = threadIdx.x;
  const float p = (float)pos[t];
  for (int idx = tid; idx < kNH * 64; idx += 256) {
    const int head = idx >> 6, i = idx & 63;
    const float ang = p * __powf(500000.0f, -(float)i * (1.0f / 64.0f));
    const float c = __cosf(ang), s = __sinf(ang);
    const size_t b = (size_t)t * (kNH * kHD) + head * kHD + i;
    const float x1 = qf[b], x2 = qf[b + 64];
    qb[b]      = f2bf(x1 * c - x2 * s);
    qb[b + 64] = f2bf(x1 * s + x2 * c);
  }
  for (int idx = tid; idx < kNKV * 64; idx += 256) {
    const int head = idx >> 6, i = idx & 63;
    const float ang = p * __powf(500000.0f, -(float)i * (1.0f / 64.0f));
    const float c = __cosf(ang), s = __sinf(ang);
    const size_t b = (size_t)t * (kNKV * kHD) + head * kHD + i;
    const float x1 = kf[b], x2 = kf[b + 64];
    kb[b]      = f2bf(x1 * c - x2 * s);
    kb[b + 64] = f2bf(x1 * s + x2 * c);
  }
  for (int i = tid; i < kNKV * kHD; i += 256) {
    const size_t b = (size_t)t * (kNKV * kHD) + i;
    vb[b] = f2bf(vf[b]);
  }
}

// ----------------------------------------------------------------- WMMA GEMM
// C[M,N] = A_bf16[M,K] @ B_f32[K,N]; B converted fp32->bf16 during LDS staging.
// Block tile 128x128, BK=32, 256 threads = 8 waves (4 m-waves x 2 n-waves,
// each wave a 32x64 tile). Software pipelined: next K-tile global loads are
// issued between the barrier and the WMMA burst.
// grid: (x = M/128  [fastest, shares the B panel -> L2 reuse], y = N/128)
// DUAL=false: Cf = A@B1 (+resid)   DUAL=true: Cb = silu(A@B1)*(A@B2) (bf16)
template <bool DUAL>
__global__ __launch_bounds__(256)
void gemm_bf16_kernel(const unsigned short* __restrict__ A,
                      const float* __restrict__ B1, const float* __restrict__ B2,
                      float* __restrict__ Cf, unsigned short* __restrict__ Cb,
                      const float* __restrict__ resid, int M, int N, int K) {
  __shared__ __align__(16) unsigned short As[128][40];   // [m][k]
  __shared__ __align__(16) unsigned short Bs1[128][40];  // [n][k] (transposed)
  __shared__ __align__(16) unsigned short Bs2[128][40];

  const int tid = threadIdx.x;
  const int lane = tid & 31, wid = tid >> 5;
  const int l16 = lane & 15, hi = lane >> 4;
  const int wm = wid >> 1, wn = wid & 1;     // 4x2 wave grid of 32x64 tiles
  const int m0 = blockIdx.x * 128;           // x = M-block (fastest varying)
  const int n0 = blockIdx.y * 128;           // y = weight panel (L2-resident)

  v8f acc1[2][4] = {};
  v8f acc2[2][4] = {};

  // staging assignments
  const int arow = tid >> 1, ac0 = (tid & 1) * 16;   // A: 128 rows x 32 k
  const int bk0 = (tid & 15) * 2, bc0 = (tid >> 4) * 8;  // B: k-pairs x 8 n

  u32x4 aReg[2];
  f32x4 b1Reg[4], b2Reg[4];

  auto load_tile = [&](int kt) {
    const unsigned short* ag = A + (size_t)(m0 + arow) * K + kt + ac0;
    aReg[0] = *(const u32x4*)(ag);
    aReg[1] = *(const u32x4*)(ag + 8);
    const float* bg1 = B1 + (size_t)(kt + bk0) * N + n0 + bc0;
    b1Reg[0] = *(const f32x4*)(bg1);
    b1Reg[1] = *(const f32x4*)(bg1 + 4);
    b1Reg[2] = *(const f32x4*)(bg1 + N);
    b1Reg[3] = *(const f32x4*)(bg1 + N + 4);
    if (DUAL) {
      const float* bg2 = B2 + (size_t)(kt + bk0) * N + n0 + bc0;
      b2Reg[0] = *(const f32x4*)(bg2);
      b2Reg[1] = *(const f32x4*)(bg2 + 4);
      b2Reg[2] = *(const f32x4*)(bg2 + N);
      b2Reg[3] = *(const f32x4*)(bg2 + N + 4);
    }
  };
  auto store_tile = [&]() {
    *(u32x4*)&As[arow][ac0]     = aReg[0];
    *(u32x4*)&As[arow][ac0 + 8] = aReg[1];
#pragma unroll
    for (int j = 0; j < 8; ++j) {
      const float lo = (j < 4) ? b1Reg[0][j] : b1Reg[1][j - 4];
      const float hv = (j < 4) ? b1Reg[2][j] : b1Reg[3][j - 4];
      *(unsigned*)&Bs1[bc0 + j][bk0] = pack_bf2(lo, hv);
      if (DUAL) {
        const float lo2 = (j < 4) ? b2Reg[0][j] : b2Reg[1][j - 4];
        const float hv2 = (j < 4) ? b2Reg[2][j] : b2Reg[3][j - 4];
        *(unsigned*)&Bs2[bc0 + j][bk0] = pack_bf2(lo2, hv2);
      }
    }
  };

  load_tile(0);
  for (int kt = 0; kt < K; kt += 32) {
    store_tile();
    __syncthreads();
    if (kt + 32 < K) load_tile(kt + 32);   // overlap next-tile loads w/ WMMA

    FragBF a[2], b1[4], b2[4];
#pragma unroll
    for (int mi = 0; mi < 2; ++mi) {
      const int r = wm * 32 + mi * 16 + l16;
      a[mi].q[0] = *(const u32x4*)&As[r][hi * 16];       // k = hi*16 + 0..7
      a[mi].q[1] = *(const u32x4*)&As[r][hi * 16 + 8];   // k = hi*16 + 8..15
    }
#pragma unroll
    for (int ni = 0; ni < 4; ++ni) {
      const int c = wn * 64 + ni * 16 + l16;
      b1[ni].q[0] = *(const u32x4*)&Bs1[c][hi * 16];     // k = hi*16 + 0..7
      b1[ni].q[1] = *(const u32x4*)&Bs1[c][hi * 16 + 8]; // k = hi*16 + 8..15
      if (DUAL) {
        b2[ni].q[0] = *(const u32x4*)&Bs2[c][hi * 16];
        b2[ni].q[1] = *(const u32x4*)&Bs2[c][hi * 16 + 8];
      }
    }
#pragma unroll
    for (int mi = 0; mi < 2; ++mi)
#pragma unroll
      for (int ni = 0; ni < 4; ++ni) {
        acc1[mi][ni] = __builtin_amdgcn_wmma_f32_16x16x32_bf16(
            false, a[mi].v, false, b1[ni].v, (short)0, acc1[mi][ni], false, false);
        if (DUAL)
          acc2[mi][ni] = __builtin_amdgcn_wmma_f32_16x16x32_bf16(
              false, a[mi].v, false, b2[ni].v, (short)0, acc2[mi][ni], false, false);
      }
    __syncthreads();
  }

#pragma unroll
  for (int mi = 0; mi < 2; ++mi)
#pragma unroll
    for (int ni = 0; ni < 4; ++ni)
#pragma unroll
      for (int r = 0; r < 8; ++r) {
        const int row = m0 + wm * 32 + mi * 16 + hi * 8 + r;
        const int col = n0 + wn * 64 + ni * 16 + l16;
        const size_t idx = (size_t)row * N + col;
        if (DUAL) {
          const float g = acc1[mi][ni][r], u = acc2[mi][ni][r];
          Cb[idx] = f2bf((g / (1.0f + __expf(-g))) * u);
        } else {
          float v = acc1[mi][ni][r];
          if (resid) v += resid[idx];
          Cf[idx] = v;
        }
      }
}

// ------------------------------------------------------- flash attention (WMMA)
// grid: (T/64, heads); block: 128 threads = 4 waves; each wave owns 16 q rows.
__global__ __launch_bounds__(128)
void attn_kernel(const unsigned short* __restrict__ Qb,
                 const unsigned short* __restrict__ Kbuf,
                 const unsigned short* __restrict__ Vbuf,
                 unsigned short* __restrict__ Ob) {
  __shared__ __align__(16) unsigned short Kt[32][136];    // [key][dim]
  __shared__ __align__(16) unsigned short Vt[128][40];    // [dim][key] transposed
  __shared__ __align__(16) unsigned short Pw[4][16][40];  // per-wave P tile

  const int q0 = blockIdx.x * 64;
  const int h = blockIdx.y;
  const int kvh = h >> 2;                    // GQA: 4 q-heads per kv head
  const int tid = threadIdx.x;
  const int lane = tid & 31, wid = tid >> 5;
  const int l16 = lane & 15, hi = lane >> 4;

  FragBF qfrag[4];
  {
    const int qrow = q0 + wid * 16 + l16;
    const unsigned short* qp = Qb + (size_t)qrow * (kNH * kHD) + h * kHD;
#pragma unroll
    for (int c = 0; c < 4; ++c) {
      qfrag[c].q[0] = *(const u32x4*)(qp + c * 32 + hi * 16);
      qfrag[c].q[1] = *(const u32x4*)(qp + c * 32 + hi * 16 + 8);
    }
  }

  float mrow[8], lrow[8];
  v8f Oacc[8] = {};
#pragma unroll
  for (int r = 0; r < 8; ++r) { mrow[r] = -3.0e38f; lrow[r] = 0.f; }

  const float scale = 0.08838834764831845f;  // 1/sqrt(128)
  const int kend = q0 + 64;
  const int skey = tid >> 2, sc0 = (tid & 3) * 32;

  for (int kblk = 0; kblk < kend; kblk += 32) {
    __syncthreads();
    {
      const unsigned short* kg =
          Kbuf + (size_t)(kblk + skey) * (kNKV * kHD) + kvh * kHD + sc0;
#pragma unroll
      for (int i = 0; i < 4; ++i)
        *(u32x4*)&Kt[skey][sc0 + i * 8] = *(const u32x4*)(kg + i * 8);
      const unsigned short* vg =
          Vbuf + (size_t)(kblk + skey) * (kNKV * kHD) + kvh * kHD + sc0;
#pragma unroll
      for (int i = 0; i < 4; ++i) {
        union { u32x4 q; unsigned short s[8]; } tmp;
        tmp.q = *(const u32x4*)(vg + i * 8);
#pragma unroll
        for (int j = 0; j < 8; ++j) Vt[sc0 + i * 8 + j][skey] = tmp.s[j];
      }
    }
    __syncthreads();

    v8f S[2] = {};
#pragma unroll
    for (int g = 0; g < 2; ++g) {
#pragma unroll
      for (int c = 0; c < 4; ++c) {
        FragBF kfr;
        kfr.q[0] = *(const u32x4*)&Kt[g * 16 + l16][c * 32 + hi * 16];
        kfr.q[1] = *(const u32x4*)&Kt[g * 16 + l16][c * 32 + hi * 16 + 8];
        S[g] = __builtin_amdgcn_wmma_f32_16x16x32_bf16(
            false, qfrag[c].v, false, kfr.v, (short)0, S[g], false, false);
      }
    }

#pragma unroll
    for (int r = 0; r < 8; ++r) {
      const int q_abs = q0 + wid * 16 + hi * 8 + r;
      float s0 = S[0][r] * scale, s1 = S[1][r] * scale;
      if (kblk + l16 > q_abs) s0 = -3.0e38f;
      if (kblk + 16 + l16 > q_abs) s1 = -3.0e38f;
      float mx = fmaxf(s0, s1);
#pragma unroll
      for (int d = 1; d < 16; d <<= 1) mx = fmaxf(mx, __shfl_xor(mx, d, 32));
      const float mn = fmaxf(mrow[r], mx);
      const float alpha = __expf(mrow[r] - mn);
      mrow[r] = mn;
      const float p0 = __expf(s0 - mn), p1 = __expf(s1 - mn);
      float ps = p0 + p1;
#pragma unroll
      for (int d = 1; d < 16; d <<= 1) ps += __shfl_xor(ps, d, 32);
      lrow[r] = lrow[r] * alpha + ps;
#pragma unroll
      for (int dd = 0; dd < 8; ++dd) Oacc[dd][r] *= alpha;
      Pw[wid][hi * 8 + r][l16] = f2bf(p0);
      Pw[wid][hi * 8 + r][16 + l16] = f2bf(p1);
    }
    wait_lds();  // per-wave LDS RAW (DS in-order per wave)

    FragBF pf;
    pf.q[0] = *(const u32x4*)&Pw[wid][l16][hi * 16];
    pf.q[1] = *(const u32x4*)&Pw[wid][l16][hi * 16 + 8];
#pragma unroll
    for (int dd = 0; dd < 8; ++dd) {
      FragBF vfr;
      vfr.q[0] = *(const u32x4*)&Vt[dd * 16 + l16][hi * 16];
      vfr.q[1] = *(const u32x4*)&Vt[dd * 16 + l16][hi * 16 + 8];
      Oacc[dd] = __builtin_amdgcn_wmma_f32_16x16x32_bf16(
          false, pf.v, false, vfr.v, (short)0, Oacc[dd], false, false);
    }
  }

#pragma unroll
  for (int dd = 0; dd < 8; ++dd)
#pragma unroll
    for (int r = 0; r < 8; ++r) {
      const int row = q0 + wid * 16 + hi * 8 + r;
      const float v = Oacc[dd][r] / lrow[r];
      Ob[(size_t)row * (kNH * kHD) + h * kHD + dd * 16 + l16] = f2bf(v);
    }
}

// ------------------------------------------------------------------- launcher
extern "C" void kernel_launch(void* const* d_in, const int* in_sizes, int n_in,
                              void* d_out, int out_size, void* d_ws, size_t ws_size,
                              hipStream_t stream) {
  const float* hidden    = (const float*)d_in[0];
  const int*   positions = (const int*)d_in[1];
  const float* w_q       = (const float*)d_in[2];
  const float* w_k       = (const float*)d_in[3];
  const float* w_v       = (const float*)d_in[4];
  const float* w_o       = (const float*)d_in[5];
  const float* norm1_w   = (const float*)d_in[6];
  const float* norm2_w   = (const float*)d_in[7];
  const float* w_gate    = (const float*)d_in[8];
  const float* w_up      = (const float*)d_in[9];
  const float* w_down    = (const float*)d_in[10];
  float* out = (float*)d_out;

  char* ws = (char*)d_ws;                                   // 160 MB layout
  unsigned short* xn    = (unsigned short*)(ws);                       // 16MB (reused as hn)
  float*          qf    = (float*)(ws + (size_t)16  * 1048576);        // 32MB (reused as x1)
  float*          kf    = (float*)(ws + (size_t)48  * 1048576);        // 8MB
  float*          vf    = (float*)(ws + (size_t)56  * 1048576);        // 8MB
  unsigned short* qb    = (unsigned short*)(ws + (size_t)64 * 1048576);// 16MB
  unsigned short* kb    = (unsigned short*)(ws + (size_t)80 * 1048576);// 4MB
  unsigned short* vb    = (unsigned short*)(ws + (size_t)84 * 1048576);// 4MB
  unsigned short* attnb = (unsigned short*)(ws + (size_t)88 * 1048576);// 16MB
  unsigned short* actb  = (unsigned short*)(ws + (size_t)104* 1048576);// 56MB

  // 1) pre-attention rmsnorm -> bf16
  rmsnorm_kernel<<<kT, 256, 0, stream>>>(hidden, norm1_w, xn);
  // 2) QKV projections (fp32 out, pre-rope); grid.x = M-blocks (L2 panel reuse)
  gemm_bf16_kernel<false><<<dim3(kT / 128, 32), 256, 0, stream>>>(
      xn, w_q, nullptr, qf, nullptr, nullptr, kT, kNH * kHD, kHidden);
  gemm_bf16_kernel<false><<<dim3(kT / 128, 8), 256, 0, stream>>>(
      xn, w_k, nullptr, kf, nullptr, nullptr, kT, kNKV * kHD, kHidden);
  gemm_bf16_kernel<false><<<dim3(kT / 128, 8), 256, 0, stream>>>(
      xn, w_v, nullptr, vf, nullptr, nullptr, kT, kNKV * kHD, kHidden);
  // 3) rope + convert to bf16
  rope_cvt_kernel<<<kT, 256, 0, stream>>>(qf, kf, vf, positions, qb, kb, vb);
  // 4) causal GQA flash attention -> bf16
  attn_kernel<<<dim3(kT / 64, kNH), 128, 0, stream>>>(qb, kb, vb, attnb);
  // 5) o-proj + residual (x1 reuses qf)
  float* x1 = qf;
  gemm_bf16_kernel<false><<<dim3(kT / 128, 32), 256, 0, stream>>>(
      attnb, w_o, nullptr, x1, nullptr, hidden, kT, kHidden, kNH * kHD);
  // 6) post-attention rmsnorm -> bf16 (reuses xn)
  rmsnorm_kernel<<<kT, 256, 0, stream>>>(x1, norm2_w, xn);
  // 7) fused gate/up dual-GEMM + silu*up -> bf16
  gemm_bf16_kernel<true><<<dim3(kT / 128, kInter / 128), 256, 0, stream>>>(
      xn, w_gate, w_up, nullptr, actb, nullptr, kT, kInter, kHidden);
  // 8) down-proj + residual -> d_out
  gemm_bf16_kernel<false><<<dim3(kT / 128, 32), 256, 0, stream>>>(
      actb, w_down, nullptr, out, nullptr, x1, kT, kHidden, kInter);
}
